// GCNEdgePrediction_48473000903025
// MI455X (gfx1250) — compile-verified, matrix operand order checked by reading, not run on hardware
//
#include <hip/hip_runtime.h>
#include <hip/hip_bf16.h>

typedef __attribute__((ext_vector_type(16))) _Float16 v16h;
typedef __attribute__((ext_vector_type(8)))  float    v8f;

__device__ __forceinline__ void atomAddF(float* p, float v) {
    // non-returning f32 atomic -> global_atomic_add_f32 (no CAS loop)
    __hip_atomic_fetch_add(p, v, __ATOMIC_RELAXED, __HIP_MEMORY_SCOPE_AGENT);
}

// ---------------- degree / normalization ----------------
__global__ void k_deg_init(float* __restrict__ deg, int n) {
    int i = blockIdx.x * blockDim.x + threadIdx.x;
    if (i < n) deg[i] = 1.0f;                       // self-loop contribution
}
__global__ void k_deg_count(const int* __restrict__ dst, float* __restrict__ deg, int e) {
    int i = blockIdx.x * blockDim.x + threadIdx.x;
    if (i < e) atomAddF(&deg[dst[i]], 1.0f);
}
__global__ void k_deg_rsqrt(float* __restrict__ deg, int n) {
    int i = blockIdx.x * blockDim.x + threadIdx.x;
    if (i < n) deg[i] = rsqrtf(deg[i]);             // deg >= 1 always
}

// ---------------- weight transpose: Wt[n*K + k] = W[k*Nout + n] ----------------
__global__ void k_transposeW(const float* __restrict__ W, float* __restrict__ Wt,
                             int K, int Nout) {
    int i = blockIdx.x * blockDim.x + threadIdx.x;
    if (i >= K * Nout) return;
    int k = i / Nout, n = i - k * Nout;
    Wt[(size_t)n * K + k] = W[i];
}

// ---------------- GEMM: C[N,Nout] = A[N,K] @ W[K,Nout], split-f16 WMMA ----------------
// Wt is W transposed ([Nout,K]) so the B-column gather is contiguous over K.
// grid = (N/16, Nout/64), block = 128 (4 waves); wave w -> tile (blockIdx.x, blockIdx.y*4+w)
__global__ void k_gemm_wmma(const float* __restrict__ A, const float* __restrict__ Wt,
                            float* __restrict__ C, int N, int K, int Nout) {
    const int lane  = threadIdx.x & 31;
    const int wid   = threadIdx.x >> 5;
    const int tileM = blockIdx.x;
    const int tileN = blockIdx.y * 4 + wid;
    const int r     = lane & 15;
    const int khalf = lane >> 4;
    const int row   = tileM * 16 + r;               // A row gathered by this lane
    const int coln  = tileN * 16 + r;               // B column gathered by this lane

    v8f acc = {};
    for (int k0 = 0; k0 < K; k0 += 32) {
        // ISA 16-bit A 16x32 layout: this lane's 16 slots are K runs
        // [k0+khalf*8, +7] (VGPR 0-3) and [k0+16+khalf*8, +7] (VGPR 4-7),
        // both contiguous -> four b128 loads each for A and B.
        const float* ap = A  + (size_t)row  * K + k0 + khalf * 8;
        const float* wp = Wt + (size_t)coln * K + k0 + khalf * 8;
        const float4 ta0 = *(const float4*)(ap);
        const float4 ta1 = *(const float4*)(ap + 4);
        const float4 ta2 = *(const float4*)(ap + 16);
        const float4 ta3 = *(const float4*)(ap + 20);
        const float4 tw0 = *(const float4*)(wp);
        const float4 tw1 = *(const float4*)(wp + 4);
        const float4 tw2 = *(const float4*)(wp + 16);
        const float4 tw3 = *(const float4*)(wp + 20);
        const float abuf[16] = {ta0.x, ta0.y, ta0.z, ta0.w, ta1.x, ta1.y, ta1.z, ta1.w,
                                ta2.x, ta2.y, ta2.z, ta2.w, ta3.x, ta3.y, ta3.z, ta3.w};
        const float wbuf[16] = {tw0.x, tw0.y, tw0.z, tw0.w, tw1.x, tw1.y, tw1.z, tw1.w,
                                tw2.x, tw2.y, tw2.z, tw2.w, tw3.x, tw3.y, tw3.z, tw3.w};
        v16h ah, al, bh, bl;
#pragma unroll
        for (int i = 0; i < 16; ++i) {
            const _Float16 h = (_Float16)abuf[i];
            ah[i] = h;
            al[i] = (_Float16)(abuf[i] - (float)h);
            const _Float16 g = (_Float16)wbuf[i];
            bh[i] = g;
            bl[i] = (_Float16)(wbuf[i] - (float)g);
        }
        // fp32-accurate product via 2-term f16 split: hi*hi + lo*hi + hi*lo
        acc = __builtin_amdgcn_wmma_f32_16x16x32_f16(false, ah, false, bh, (short)0, acc, false, false);
        acc = __builtin_amdgcn_wmma_f32_16x16x32_f16(false, al, false, bh, (short)0, acc, false, false);
        acc = __builtin_amdgcn_wmma_f32_16x16x32_f16(false, ah, false, bl, (short)0, acc, false, false);
    }
    // D layout: lanes 0-15 N=lane, M=v; lanes 16-31 N=lane-16, M=8+v
#pragma unroll
    for (int v = 0; v < 8; ++v) {
        const int m = khalf * 8 + v;
        C[(size_t)(tileM * 16 + m) * Nout + coln] = acc[v];
    }
}

// ---------------- aggregation ----------------
// init agg with self-loop term: agg[i] = T[i] * dinv[i]^2  (one wave per row, float4 lanes)
__global__ void k_selfloop(const float* __restrict__ T, const float* __restrict__ dinv,
                           float* __restrict__ agg, int n) {
    int tid = blockIdx.x * blockDim.x + threadIdx.x;
    int row = tid >> 5;
    if (row >= n) return;
    int c = (tid & 31) * 4;
    float di = dinv[row];
    float w  = di * di;
    const float4 v = *(const float4*)(T + (size_t)row * 128 + c);
    float4 o; o.x = v.x * w; o.y = v.y * w; o.z = v.z * w; o.w = v.w * w;
    *(float4*)(agg + (size_t)row * 128 + c) = o;
}

// one wave per edge: coalesced 512B row gather, scatter-add 128 floats to dst row
__global__ void k_scatter(const float* __restrict__ T, const int* __restrict__ src,
                          const int* __restrict__ dst, const float* __restrict__ dinv,
                          float* __restrict__ agg, int e) {
    int tid = blockIdx.x * blockDim.x + threadIdx.x;
    int ed  = tid >> 5;
    if (ed >= e) return;
    int c = (tid & 31) * 4;
    int s = src[ed], d = dst[ed];
    float nrm = dinv[s] * dinv[d];
    const float4 v = *(const float4*)(T + (size_t)s * 128 + c);
    float* o = agg + (size_t)d * 128 + c;
    atomAddF(o + 0, v.x * nrm);
    atomAddF(o + 1, v.y * nrm);
    atomAddF(o + 2, v.z * nrm);
    atomAddF(o + 3, v.w * nrm);
}

__global__ void k_bias_act(float* __restrict__ h, const float* __restrict__ b,
                           long long total, int relu) {
    long long i = blockIdx.x * (long long)blockDim.x + threadIdx.x;
    if (i >= total) return;
    float v = h[i] + b[i & 127];
    h[i] = relu ? fmaxf(v, 0.0f) : v;
}

// ---------------- decode: logits[e] = dot(z[a], z[b]) over 128 ----------------
__global__ void k_decode(const float* __restrict__ z, const int* __restrict__ eli,
                         float* __restrict__ out, int el) {
    int tid = blockIdx.x * blockDim.x + threadIdx.x;
    int e   = tid >> 5;
    if (e >= el) return;
    int lane = tid & 31;
    int a = eli[e], b = eli[el + e];
    const float4 va = *(const float4*)(z + (size_t)a * 128 + lane * 4);
    const float4 vb = *(const float4*)(z + (size_t)b * 128 + lane * 4);
    float s = va.x * vb.x + va.y * vb.y + va.z * vb.z + va.w * vb.w;
#pragma unroll
    for (int off = 16; off > 0; off >>= 1) s += __shfl_down(s, off, 32);
    if (lane == 0) out[e] = s;
}

extern "C" void kernel_launch(void* const* d_in, const int* in_sizes, int n_in,
                              void* d_out, int out_size, void* d_ws, size_t ws_size,
                              hipStream_t stream) {
    const float* x   = (const float*)d_in[0];
    const int*   ei  = (const int*)d_in[1];
    const int*   eli = (const int*)d_in[2];
    const float* W0  = (const float*)d_in[3];
    const float* b0  = (const float*)d_in[4];
    const float* W1  = (const float*)d_in[5];
    const float* b1  = (const float*)d_in[6];
    const float* W2  = (const float*)d_in[7];
    const float* b2  = (const float*)d_in[8];

    const int H   = 128;
    const int Fin = in_sizes[3] / H;          // 256
    const int N   = in_sizes[0] / Fin;        // 50000 (multiple of 16)
    const int E   = in_sizes[1] / 2;          // 320000
    const int EL  = in_sizes[2] / 2;          // 200000
    const int* src = ei;
    const int* dst = ei + E;

    float* ws   = (float*)d_ws;
    float* dinv = ws;                                         // N floats
    size_t off  = ((size_t)N + 255) & ~(size_t)255;
    float* Wt   = ws + off;                                   // up to Fin*H floats (transposed weights)
    float* T    = Wt + (size_t)Fin * H;                       // N*128 (GEMM out)
    float* Hb   = T + (size_t)N * H;                          // N*128 (aggregate out)

    const int TB = 256;
    const long long totalNH = (long long)N * H;
    const int gRow   = (int)((totalNH / 4 + TB - 1) / TB);    // one wave per row, f4 lanes
    const int gElem  = (int)((totalNH + TB - 1) / TB);
    const int gScat  = (int)(((long long)E * 32 + TB - 1) / TB);
    const int gDec   = (int)(((long long)EL * 32 + TB - 1) / TB);
    const dim3 gG(N / 16, (H / 16) / 4);                      // (3125, 2), 4 waves/block

    // symmetric normalization coefficients
    k_deg_init <<<(N + TB - 1) / TB, TB, 0, stream>>>(dinv, N);
    k_deg_count<<<(E + TB - 1) / TB, TB, 0, stream>>>(dst, dinv, E);
    k_deg_rsqrt<<<(N + TB - 1) / TB, TB, 0, stream>>>(dinv, N);

    // layer 0: T = x@W0 ; Hb = aggregate(T) ; Hb = relu(Hb + b0)
    k_transposeW<<<(Fin * H + TB - 1) / TB, TB, 0, stream>>>(W0, Wt, Fin, H);
    k_gemm_wmma<<<gG, 128, 0, stream>>>(x, Wt, T, N, Fin, H);
    k_selfloop <<<gRow, TB, 0, stream>>>(T, dinv, Hb, N);
    k_scatter  <<<gScat, TB, 0, stream>>>(T, src, dst, dinv, Hb, E);
    k_bias_act <<<gElem, TB, 0, stream>>>(Hb, b0, totalNH, 1);

    // layer 1
    k_transposeW<<<(H * H + TB - 1) / TB, TB, 0, stream>>>(W1, Wt, H, H);
    k_gemm_wmma<<<gG, 128, 0, stream>>>(Hb, Wt, T, N, H, H);
    k_selfloop <<<gRow, TB, 0, stream>>>(T, dinv, Hb, N);
    k_scatter  <<<gScat, TB, 0, stream>>>(T, src, dst, dinv, Hb, E);
    k_bias_act <<<gElem, TB, 0, stream>>>(Hb, b1, totalNH, 1);

    // layer 2 (no relu)
    k_transposeW<<<(H * H + TB - 1) / TB, TB, 0, stream>>>(W2, Wt, H, H);
    k_gemm_wmma<<<gG, 128, 0, stream>>>(Hb, Wt, T, N, H, H);
    k_selfloop <<<gRow, TB, 0, stream>>>(T, dinv, Hb, N);
    k_scatter  <<<gScat, TB, 0, stream>>>(T, src, dst, dinv, Hb, E);
    k_bias_act <<<gElem, TB, 0, stream>>>(Hb, b2, totalNH, 0);

    // decode
    k_decode   <<<gDec, TB, 0, stream>>>(Hb, eli, (float*)d_out, EL);
}
